// Attention_69277822485178
// MI455X (gfx1250) — matrix operand order repaired and checked by reading, not compiled, not run on hardware
//
#include <hip/hip_runtime.h>
#include <hip/hip_bf16.h>

// Problem dims
#define NB 32
#define LQ 256
#define DM 1024
#define NH 16
#define DHD 64

typedef __attribute__((ext_vector_type(16))) __bf16 v16bf;
typedef __attribute__((ext_vector_type(8)))  float  v8f;

struct Bits256 { uint4 lo; uint4 hi; };

__device__ __forceinline__ __bf16 f2bf(float f) {
  unsigned u = __builtin_bit_cast(unsigned, f);
  u += 0x7FFFu + ((u >> 16) & 1u);           // round-to-nearest-even
  unsigned short h = (unsigned short)(u >> 16);
  return __builtin_bit_cast(__bf16, h);
}

__device__ __forceinline__ unsigned pack2bf(float a, float b) {
  unsigned ua = __builtin_bit_cast(unsigned, a);
  ua = (ua + 0x7FFFu + ((ua >> 16) & 1u)) >> 16;
  unsigned ub = __builtin_bit_cast(unsigned, b);
  ub = (ub + 0x7FFFu + ((ub >> 16) & 1u)) & 0xFFFF0000u;
  return ua | ub;
}

__device__ __forceinline__ v8f wmma_bf16(v16bf a, v16bf b, v8f c) {
  return __builtin_amdgcn_wmma_f32_16x16x32_bf16(false, a, false, b, (short)0, c,
                                                 false, false);
}

__device__ __forceinline__ void wait_async0() {
#if __has_builtin(__builtin_amdgcn_s_wait_asynccnt)
  __builtin_amdgcn_s_wait_asynccnt(0);
#else
  asm volatile("s_wait_asynccnt 0x0" ::: "memory");
#endif
}

// A fragment (16x32, bf16 row-major source, row stride ld elems).
// Lane m = lane&15; K split: VGPR0-3 -> kh+0..7, VGPR4-7 -> kh+16..23, kh = 8*(lane>>4).
__device__ __forceinline__ v16bf load_a_bf16(const __bf16* p, int ld) {
  int lane = threadIdx.x & 31;
  int m = lane & 15;
  int kh = (lane >> 4) << 3;
  const __bf16* r = p + (size_t)m * ld + kh;
  Bits256 bits;
  bits.lo = *(const uint4*)(r);
  bits.hi = *(const uint4*)(r + 16);
  return __builtin_bit_cast(v16bf, bits);
}

// B fragment (32x16 = K x N) where the source holds the *transpose* row-major:
// src[n][k]. Lane n = lane&15; K split: VGPR0-3 -> kh+0..7, VGPR4-7 -> kh+16..23, kh = 8*(lane>>4).
__device__ __forceinline__ v16bf load_b_bf16(const __bf16* p, int ld) {
  int lane = threadIdx.x & 31;
  int n = lane & 15;
  int kh = (lane >> 4) << 3;
  const __bf16* r = p + (size_t)n * ld + kh;
  Bits256 bits;
  bits.lo = *(const uint4*)(r);
  bits.hi = *(const uint4*)(r + 16);
  return __builtin_bit_cast(v16bf, bits);
}

// ---------------------------------------------------------------------------
// Zero the probs-mean output region (accumulated by atomics in attn_kernel).
__global__ void zero_kernel(float* __restrict__ p, unsigned long long n) {
  unsigned long long i = (unsigned long long)blockIdx.x * blockDim.x + threadIdx.x;
  if (i < n) p[i] = 0.0f;
}

// ---------------------------------------------------------------------------
// One-shot fp32 -> bf16 conversion (8 elems/thread, 16B stores).
__global__ __launch_bounds__(256) void conv_bf16_kernel(const float* __restrict__ src,
                                                        __bf16* __restrict__ dst) {
  unsigned i = (blockIdx.x * 256u + threadIdx.x) * 8u;
  float4 f0 = *(const float4*)(src + i);
  float4 f1 = *(const float4*)(src + i + 4);
  uint4 o;
  o.x = pack2bf(f0.x, f0.y);
  o.y = pack2bf(f0.z, f0.w);
  o.z = pack2bf(f1.x, f1.y);
  o.w = pack2bf(f1.z, f1.w);
  *(uint4*)(dst + i) = o;
}

// ---------------------------------------------------------------------------
// Projection: Y(bf16, MxN) = X(bf16, MxK) @ W^T(bf16, NxK) + bias(f32).
// Block tile 128(M) x 64(N); 8 waves arranged 4x2, each wave 32x32 (2x2 tiles).
__global__ __launch_bounds__(256) void proj_kernel(const __bf16* __restrict__ X,
                                                   const __bf16* __restrict__ W,
                                                   const float* __restrict__ bias,
                                                   __bf16* __restrict__ Y) {
  const int bm = blockIdx.x;           // M/128 = 64
  const int bn = blockIdx.y;           // N/64  = 16
  const int w = threadIdx.x >> 5;
  const int lane = threadIdx.x & 31;
  const int wm = w >> 1, wn = w & 1;
  const int row0 = bm * 128 + wm * 32;
  const int col0 = bn * 64 + wn * 32;

  v8f acc[2][2] = {};
  for (int kk = 0; kk < DM; kk += 32) {
    v16bf a0 = load_a_bf16(X + (size_t)row0 * DM + kk, DM);
    v16bf a1 = load_a_bf16(X + (size_t)(row0 + 16) * DM + kk, DM);
    v16bf b0 = load_b_bf16(W + (size_t)col0 * DM + kk, DM);
    v16bf b1 = load_b_bf16(W + (size_t)(col0 + 16) * DM + kk, DM);
    acc[0][0] = wmma_bf16(a0, b0, acc[0][0]);
    acc[0][1] = wmma_bf16(a0, b1, acc[0][1]);
    acc[1][0] = wmma_bf16(a1, b0, acc[1][0]);
    acc[1][1] = wmma_bf16(a1, b1, acc[1][1]);
  }
  const int lh = lane >> 4, n16 = lane & 15;
#pragma unroll
  for (int i = 0; i < 2; ++i)
#pragma unroll
    for (int j = 0; j < 2; ++j) {
      int c = col0 + j * 16 + n16;
      float bv = bias[c];
#pragma unroll
      for (int v = 0; v < 8; ++v) {
        int rr = row0 + i * 16 + v + 8 * lh;
        Y[(size_t)rr * DM + c] = f2bf(acc[i][j][v] + bv);
      }
    }
}

// ---------------------------------------------------------------------------
// Attention: one WG per (n, h, half). 8 waves, each owns 16 query rows.
#define S_LD 260   // padded f32 stride for raw scores (128x260 = 133,120 B)
#define PB_LD 264  // padded bf16 stride for normalized probs (128x264 = 67,584 B)
#define VT_LD 264  // padded bf16 stride for transposed V (64x264 = 33,792 B)

__global__ __launch_bounds__(256) void attn_kernel(const __bf16* __restrict__ Qb,
                                                   const __bf16* __restrict__ Kb,
                                                   const __bf16* __restrict__ Vb,
                                                   float* __restrict__ ctx_out,
                                                   float* __restrict__ probs_out) {
  __shared__ float  S[128 * S_LD];
  __shared__ __bf16 Pb[128 * PB_LD];
  __shared__ __bf16 Vt[DHD * VT_LD];

  const int blk = blockIdx.x;           // N*H*2 = 1024
  const int half = blk & 1;
  const int h = (blk >> 1) & (NH - 1);
  const int n = blk >> 5;
  const int l0 = half * 128;
  const int w = threadIdx.x >> 5;
  const int lane = threadIdx.x & 31;
  const int lh = lane >> 4, n16 = lane & 15;

  // Stage V_h transposed: Vt[d][m] = Vb[n, m, h*64+d]
  const __bf16* Vh = Vb + (size_t)n * LQ * DM + h * DHD;
  for (int idx = threadIdx.x; idx < DHD * LQ; idx += 256) {
    int d = idx >> 8;
    int m = idx & (LQ - 1);
    Vt[d * VT_LD + m] = Vh[(size_t)m * DM + d];
  }

  // Scores for this wave's 16 rows, all 16 column tiles, K = 64 (2 WMMA steps).
  const __bf16* Qh = Qb + (size_t)n * LQ * DM + h * DHD;
  const __bf16* Kh = Kb + (size_t)n * LQ * DM + h * DHD;
  const int rowt = l0 + w * 16;
  v16bf qa0 = load_a_bf16(Qh + (size_t)rowt * DM + 0, DM);
  v16bf qa1 = load_a_bf16(Qh + (size_t)rowt * DM + 32, DM);
  for (int ct = 0; ct < 16; ++ct) {
    v16bf kb0 = load_b_bf16(Kh + (size_t)(ct * 16) * DM + 0, DM);
    v16bf kb1 = load_b_bf16(Kh + (size_t)(ct * 16) * DM + 32, DM);
    v8f acc = {};
    acc = wmma_bf16(qa0, kb0, acc);
    acc = wmma_bf16(qa1, kb1, acc);
#pragma unroll
    for (int v = 0; v < 8; ++v)
      S[(w * 16 + v + 8 * lh) * S_LD + ct * 16 + n16] = acc[v] * 0.125f;
  }
  __syncthreads();

  // Row softmax (128 rows): write normalized probs as bf16 to Pb and
  // accumulate probs-mean (over heads) into d_out with atomics.
  if (threadIdx.x < 128) {
    float* row = &S[threadIdx.x * S_LD];
    __bf16* prow = &Pb[threadIdx.x * PB_LD];
    float mx = -3.0e38f;
    for (int m = 0; m < LQ; ++m) mx = fmaxf(mx, row[m]);
    float s = 0.0f;
    for (int m = 0; m < LQ; ++m) { float e = __expf(row[m] - mx); row[m] = e; s += e; }
    float inv = 1.0f / s;
    float* pout = probs_out + (size_t)n * LQ * LQ + (size_t)(l0 + threadIdx.x) * LQ;
    for (int m = 0; m < LQ; ++m) {
      float p = row[m] * inv;
      prow[m] = f2bf(p);
      atomicAdd(&pout[m], p * (1.0f / NH));
    }
  }
  __syncthreads();

  // ctx_h = P (128x256) @ V_h (256x64); wave: 16 rows x 4 col tiles, K = 256.
  v8f acc2[4] = {};
  for (int ks = 0; ks < LQ; ks += 32) {
    v16bf pa = load_a_bf16(&Pb[(w * 16) * PB_LD + ks], PB_LD);
#pragma unroll
    for (int ct = 0; ct < 4; ++ct) {
      v16bf vb = load_b_bf16(&Vt[(ct * 16) * VT_LD + ks], VT_LD);
      acc2[ct] = wmma_bf16(pa, vb, acc2[ct]);
    }
  }
#pragma unroll
  for (int ct = 0; ct < 4; ++ct)
#pragma unroll
    for (int v = 0; v < 8; ++v) {
      int l = l0 + w * 16 + v + 8 * lh;
      int d = h * DHD + ct * 16 + n16;
      ctx_out[(size_t)n * LQ * DM + (size_t)l * DM + d] = acc2[ct][v];
    }
}

// ---------------------------------------------------------------------------
// Residual + LayerNorm, one WG (256 threads) per (n,l) row of 1024 elems.
__global__ __launch_bounds__(256) void ln_kernel(float* __restrict__ ctx,
                                                 const float* __restrict__ resid,
                                                 const float* __restrict__ ln_w,
                                                 const float* __restrict__ ln_b) {
  __shared__ float red[256], red2[256];
  const int row = blockIdx.x;
  float* c = ctx + (size_t)row * DM;
  const float* q = resid + (size_t)row * DM;
  float x[4];
  float s = 0.0f, s2 = 0.0f;
#pragma unroll
  for (int i = 0; i < 4; ++i) {
    int idx = threadIdx.x + i * 256;
    float v = c[idx] + q[idx];
    x[i] = v; s += v; s2 += v * v;
  }
  red[threadIdx.x] = s; red2[threadIdx.x] = s2;
  __syncthreads();
  for (int off = 128; off > 0; off >>= 1) {
    if (threadIdx.x < off) {
      red[threadIdx.x]  += red[threadIdx.x + off];
      red2[threadIdx.x] += red2[threadIdx.x + off];
    }
    __syncthreads();
  }
  float mean = red[0] * (1.0f / DM);
  float var = red2[0] * (1.0f / DM) - mean * mean;
  float inv = rsqrtf(var + 1e-6f);
#pragma unroll
  for (int i = 0; i < 4; ++i) {
    int idx = threadIdx.x + i * 256;
    c[idx] = (x[i] - mean) * inv * ln_w[idx] + ln_b[idx];
  }
}

// ---------------------------------------------------------------------------
// Retrieval: one WG per (a,b); R = Qf[a](256x1024) @ Kf[b]^T, then
// t2v = sum_l max_m R, v2t = sum_m max_l R, combined with exp(logit_scale).
// The 64-row K panel for each column block is staged into LDS once via
// CDNA5 async global->LDS loads (ASYNCcnt tracked), removing the 8x redundant
// global reads across waves.
#define R_LD 68    // padded f32 stride (256x68 = 69,632 B)
#define KB_LD 1032 // padded bf16 stride for staged K panel (64x1032 = 132,096 B)

__global__ __launch_bounds__(256) void retr_kernel(const __bf16* __restrict__ Qb,
                                                   const __bf16* __restrict__ Kb,
                                                   const float* __restrict__ logit_scale,
                                                   float* __restrict__ out) {
  __shared__ __bf16 Kblk[64 * KB_LD];
  __shared__ float Rs[256 * R_LD];
  __shared__ float rowmax[256];
  __shared__ float s_t2v, s_v2t;

  const int a = blockIdx.x >> 5, b = blockIdx.x & 31;
  const __bf16* Qa = Qb + (size_t)a * LQ * DM;
  const __bf16* Kv = Kb + (size_t)b * LQ * DM;
  const int w = threadIdx.x >> 5;
  const int lane = threadIdx.x & 31;
  const int lh = lane >> 4, n16 = lane & 15;

  if (threadIdx.x == 0) { s_t2v = 0.0f; s_v2t = 0.0f; }
  rowmax[threadIdx.x] = -3.0e38f;
  __syncthreads();

  const int r0 = w * 32;                         // wave owns 32 rows
  for (int nb = 0; nb < 4; ++nb) {               // 4 column blocks of 64
    const int m0 = nb * 64;

    // Async-stage K rows [m0, m0+64) x 1024 into LDS (b128 = 8 bf16 per op).
    for (int c = threadIdx.x; c < 64 * 128; c += 256) {
      int row = c >> 7;
      int col8 = (c & 127) << 3;
      const __bf16* g = Kv + (size_t)(m0 + row) * DM + col8;
      unsigned lds_off = (unsigned)(size_t)(&Kblk[row * KB_LD + col8]);
      unsigned long long ga = (unsigned long long)(size_t)g;
      asm volatile("global_load_async_to_lds_b128 %0, %1, off"
                   :: "v"(lds_off), "v"(ga) : "memory");
    }
    wait_async0();
    __syncthreads();

    v8f acc[2][4] = {};
    for (int kk = 0; kk < DM; kk += 32) {
      v16bf a0 = load_a_bf16(Qa + (size_t)r0 * DM + kk, DM);
      v16bf a1 = load_a_bf16(Qa + (size_t)(r0 + 16) * DM + kk, DM);
#pragma unroll
      for (int ct = 0; ct < 4; ++ct) {
        v16bf bb = load_b_bf16(&Kblk[(ct * 16) * KB_LD + kk], KB_LD);
        acc[0][ct] = wmma_bf16(a0, bb, acc[0][ct]);
        acc[1][ct] = wmma_bf16(a1, bb, acc[1][ct]);
      }
    }
#pragma unroll
    for (int i = 0; i < 2; ++i)
#pragma unroll
      for (int ct = 0; ct < 4; ++ct)
#pragma unroll
        for (int v = 0; v < 8; ++v)
          Rs[(r0 + i * 16 + v + 8 * lh) * R_LD + ct * 16 + n16] = acc[i][ct][v];
    __syncthreads();

    {   // running row max over this 64-column block
      const float* rr = &Rs[threadIdx.x * R_LD];
      float mx = -3.0e38f;
      for (int m = 0; m < 64; ++m) mx = fmaxf(mx, rr[m]);
      rowmax[threadIdx.x] = fmaxf(rowmax[threadIdx.x], mx);
    }
    if (threadIdx.x < 64) {   // column max (final for these 64 cols) -> v2t sum
      float mx = -3.0e38f;
      for (int l = 0; l < LQ; ++l) mx = fmaxf(mx, Rs[l * R_LD + threadIdx.x]);
      atomicAdd(&s_v2t, mx);
    }
    __syncthreads();
  }

  atomicAdd(&s_t2v, rowmax[threadIdx.x]);
  __syncthreads();
  if (threadIdx.x == 0) {
    float ls = __expf(logit_scale[0]);
    out[a * NB + b] = ls * (s_t2v * (1.0f / LQ) + s_v2t * (1.0f / LQ)) * 0.5f;
  }
}

// ---------------------------------------------------------------------------
extern "C" void kernel_launch(void* const* d_in, const int* in_sizes, int n_in,
                              void* d_out, int out_size, void* d_ws, size_t ws_size,
                              hipStream_t stream) {
  (void)in_sizes; (void)n_in; (void)out_size; (void)ws_size;

  const float* q_in = (const float*)d_in[0];
  const float* k_in = (const float*)d_in[1];
  const float* v_in = (const float*)d_in[2];
  const float* Wq   = (const float*)d_in[3];
  const float* bq   = (const float*)d_in[4];
  const float* Wk   = (const float*)d_in[5];
  const float* bk   = (const float*)d_in[6];
  const float* Wv   = (const float*)d_in[7];
  const float* bv   = (const float*)d_in[8];
  const float* lw   = (const float*)d_in[9];
  const float* lb   = (const float*)d_in[10];
  const float* lsc  = (const float*)d_in[11];

  float* out   = (float*)d_out;
  float* ctx   = out;                                  // N*L*D
  float* ret   = out + (size_t)NB * LQ * DM;           // N*N
  float* probs = ret + (size_t)NB * NB;                // N*L*L

  const size_t NX = (size_t)NB * LQ * DM;              // 8,388,608
  const size_t NW = (size_t)DM * DM;                   // 1,048,576

  __bf16* Qb  = (__bf16*)d_ws;                         // projected Q/K/V (bf16)
  __bf16* Kbp = Qb + NX;
  __bf16* Vbp = Kbp + NX;
  __bf16* Xq  = Vbp + NX;                              // bf16 copies of inputs
  __bf16* Xk  = Xq + NX;
  __bf16* Xv  = Xk + NX;
  __bf16* Wqb = Xv + NX;                               // bf16 copies of weights
  __bf16* Wkb = Wqb + NW;
  __bf16* Wvb = Wkb + NW;

  const unsigned long long nprob = (unsigned long long)NB * LQ * LQ;
  zero_kernel<<<(unsigned)((nprob + 255) / 256), 256, 0, stream>>>(probs, nprob);

  conv_bf16_kernel<<<(unsigned)(NX / (8 * 256)), 256, 0, stream>>>(q_in, Xq);
  conv_bf16_kernel<<<(unsigned)(NX / (8 * 256)), 256, 0, stream>>>(k_in, Xk);
  conv_bf16_kernel<<<(unsigned)(NX / (8 * 256)), 256, 0, stream>>>(v_in, Xv);
  conv_bf16_kernel<<<(unsigned)(NW / (8 * 256)), 256, 0, stream>>>(Wq, Wqb);
  conv_bf16_kernel<<<(unsigned)(NW / (8 * 256)), 256, 0, stream>>>(Wk, Wkb);
  conv_bf16_kernel<<<(unsigned)(NW / (8 * 256)), 256, 0, stream>>>(Wv, Wvb);

  dim3 pg(64, 16);
  proj_kernel<<<pg, 256, 0, stream>>>(Xq, Wqb, bq, Qb);
  proj_kernel<<<pg, 256, 0, stream>>>(Xk, Wkb, bk, Kbp);
  proj_kernel<<<pg, 256, 0, stream>>>(Xv, Wvb, bv, Vbp);

  attn_kernel<<<NB * NH * 2, 256, 0, stream>>>(Qb, Kbp, Vbp, ctx, probs);
  ln_kernel<<<NB * LQ, 256, 0, stream>>>(ctx, q_in, lw, lb);
  retr_kernel<<<NB * NB, 256, 0, stream>>>(Qb, Kbp, lsc, ret);
}
